// GNN_80032420594054
// MI455X (gfx1250) — compile-verified, matrix operand order checked by reading, not run on hardware
//
#include <hip/hip_runtime.h>
#include <hip/hip_bf16.h>
#include <math.h>

#define Nn    50000
#define Ee    800000
#define ET    (Ee + Nn)     // edges + self loops
#define FIN   128
#define HIDC  64
#define Gg    512
#define NEG   0.2f

typedef __attribute__((ext_vector_type(16))) _Float16 v16h;
typedef __attribute__((ext_vector_type(8)))  _Float16 v8h;
typedef __attribute__((ext_vector_type(8)))  float    v8f;

// ---------- helpers ----------
__device__ __forceinline__ unsigned encf(float f) {
    unsigned u = __float_as_uint(f);
    return (u & 0x80000000u) ? ~u : (u | 0x80000000u);
}
__device__ __forceinline__ float decf(unsigned u) {
    u = (u & 0x80000000u) ? (u ^ 0x80000000u) : ~u;
    return __uint_as_float(u);
}

__global__ void fill_f(float* p, float v, int n) {
    int i = blockIdx.x * blockDim.x + threadIdx.x;
    if (i < n) p[i] = v;
}
__global__ void fill_u(unsigned* p, unsigned v, int n) {
    int i = blockIdx.x * blockDim.x + threadIdx.x;
    if (i < n) p[i] = v;
}

// ---------- fp32 -> fp16 convert (row-major copy) ----------
__global__ void cvt_f16(const float* __restrict__ src, _Float16* __restrict__ dst, int n) {
    int i = blockIdx.x * blockDim.x + threadIdx.x;
    if (i < n) dst[i] = (_Float16)src[i];
}

// ---------- W [K x 64] f32 -> Wt [64 x K] f16 (transpose) ----------
__global__ void transpose_w(const float* __restrict__ W, _Float16* __restrict__ Wt, int K) {
    int i = blockIdx.x * blockDim.x + threadIdx.x;
    if (i >= K * HIDC) return;
    int k = i / HIDC, c = i - k * HIDC;
    Wt[(size_t)c * K + k] = (_Float16)W[i];
}

// ---------- tiny precompute: c_e = dot(We[1x64], ae[64]) ----------
__global__ void ce_kernel(const float* __restrict__ We, const float* __restrict__ ae,
                          float* __restrict__ dst) {
    __shared__ float sh[HIDC];
    int t = threadIdx.x;
    sh[t] = We[t] * ae[t];
    __syncthreads();
    if (t == 0) {
        float s = 0.f;
        for (int i = 0; i < HIDC; ++i) s += sh[i];
        dst[0] = s;
    }
}

// ---------- sum of edge weights (for self-loop fill 'mean') ----------
__global__ void ewsum_kernel(const float* __restrict__ ew, float* __restrict__ dst) {
    int i = blockIdx.x * blockDim.x + threadIdx.x;
    float v = (i < Ee) ? ew[i] : 0.f;
    for (int o = 16; o; o >>= 1) v += __shfl_xor(v, o, 32);
    if ((threadIdx.x & 31) == 0) atomicAdd(dst, v);
}

// ---------- WMMA GEMM: H[N x 64] = Xh[N x K](f16) * Wt[64 x K](f16, pre-transposed) ----
// block = 128 threads (4 waves); wave w computes cols [16w,16w+16) of a 16-row tile.
// Fragment runs are contiguous in memory (ISA 16-bit A layout): lane (m,hi) needs
// k in [hi*8, hi*8+8) and [16+hi*8, 16+hi*8+8) -> two global_load_b128 per matrix.
template <int K>
__global__ __launch_bounds__(128)
void gemm_wmma(const _Float16* __restrict__ Xh, const _Float16* __restrict__ Wt,
               float* __restrict__ H) {
    const int lane = threadIdx.x & 31;
    const int wave = threadIdx.x >> 5;
    const int row0 = blockIdx.x * 16;
    const int hi = lane >> 4;
    const int mn = lane & 15;

    const _Float16* pa = Xh + (size_t)(row0 + mn) * K + hi * 8;
    const _Float16* pb = Wt + (size_t)(wave * 16 + mn) * K + hi * 8;

    v8f c = {};
    #pragma unroll
    for (int k0 = 0; k0 < K; k0 += 32) {
        v8h a0 = *(const v8h*)(pa + k0);
        v8h a1 = *(const v8h*)(pa + k0 + 16);
        v8h b0 = *(const v8h*)(pb + k0);
        v8h b1 = *(const v8h*)(pb + k0 + 16);
        v16h a = __builtin_shufflevector(a0, a1, 0,1,2,3,4,5,6,7,8,9,10,11,12,13,14,15);
        v16h b = __builtin_shufflevector(b0, b1, 0,1,2,3,4,5,6,7,8,9,10,11,12,13,14,15);
        c = __builtin_amdgcn_wmma_f32_16x16x32_f16(false, a, false, b,
                                                   (short)0, c, false, false);
    }
    #pragma unroll
    for (int v = 0; v < 8; ++v) {
        int m = v + hi * 8;                                   // ISA f32 C/D layout
        H[(size_t)(row0 + m) * HIDC + wave * 16 + mn] = c[v];
    }
}

// ---------- per-node attention scores: a_src=h.as, a_dst=h.ad (wave per node) ----------
__global__ void node_scores(const float* __restrict__ h, const float* __restrict__ as,
                            const float* __restrict__ ad, float* __restrict__ aSrc,
                            float* __restrict__ aDst) {
    int node = (blockIdx.x * blockDim.x + threadIdx.x) >> 5;
    int lane = threadIdx.x & 31;
    if (node >= Nn) return;
    float h0 = h[(size_t)node * HIDC + lane];
    float h1 = h[(size_t)node * HIDC + 32 + lane];
    float vs = h0 * as[lane] + h1 * as[32 + lane];
    float vd = h0 * ad[lane] + h1 * ad[32 + lane];
    for (int o = 16; o; o >>= 1) { vs += __shfl_xor(vs, o, 32); vd += __shfl_xor(vd, o, 32); }
    if (lane == 0) { aSrc[node] = vs; aDst[node] = vd; }
}

// ---------- edge pass 1: logits + leaky relu + segment max ----------
__global__ void edge_logit_max(const int* __restrict__ ei, const float* __restrict__ ew,
                               const float* __restrict__ aSrc, const float* __restrict__ aDst,
                               const float* __restrict__ ce, const float* __restrict__ ews,
                               float* __restrict__ alpha, unsigned* __restrict__ mEnc) {
    int e = blockIdx.x * blockDim.x + threadIdx.x;
    if (e >= ET) return;
    int s, d; float ea;
    if (e < Ee) { s = ei[e]; d = ei[Ee + e]; ea = ew[e]; }
    else        { s = e - Ee; d = s; ea = ews[0] * (1.0f / (float)Ee); }
    float al = aSrc[s] + aDst[d] + ea * ce[0];
    al = (al > 0.f) ? al : NEG * al;
    alpha[e] = al;
    atomicMax(mEnc + d, encf(al));
}

// ---------- edge pass 2: exp(alpha - max) + segment sum ----------
__global__ void edge_exp_sum(const int* __restrict__ ei, const unsigned* __restrict__ mEnc,
                             float* __restrict__ alpha, float* __restrict__ sSum) {
    int e = blockIdx.x * blockDim.x + threadIdx.x;
    if (e >= ET) return;
    int d = (e < Ee) ? ei[Ee + e] : (e - Ee);
    float a = expf(alpha[e] - decf(mEnc[d]));
    alpha[e] = a;
    atomicAdd(sSum + d, a);
}

// ---------- edge pass 3: weighted scatter-add (wave per edge, 2 feats/lane) ----------
__global__ void scatter_edges(const int* __restrict__ ei, const float* __restrict__ h,
                              const float* __restrict__ alpha, const float* __restrict__ sSum,
                              float* __restrict__ acc) {
    int e    = (blockIdx.x * blockDim.x + threadIdx.x) >> 5;
    int lane = threadIdx.x & 31;
    if (e >= ET) return;
    int s, d;
    if (e < Ee) { s = ei[e]; d = ei[Ee + e]; } else { s = e - Ee; d = s; }
    float w = alpha[e] / (sSum[d] + 1e-16f);
    float v0 = h[(size_t)s * HIDC + lane] * w;
    float v1 = h[(size_t)s * HIDC + 32 + lane] * w;
    atomicAdd(acc + (size_t)d * HIDC + lane, v0);
    atomicAdd(acc + (size_t)d * HIDC + 32 + lane, v1);
}

// ---------- bias + optional relu; write f16 (next-layer GEMM input) and/or f32 ----------
__global__ void finalize(const float* __restrict__ acc, const float* __restrict__ bias,
                         _Float16* __restrict__ xh, float* __restrict__ xf, int relu) {
    int i = blockIdx.x * blockDim.x + threadIdx.x;
    if (i >= Nn * HIDC) return;
    float v = acc[i] + bias[i & (HIDC - 1)];
    if (relu) v = fmaxf(v, 0.f);
    if (xh) xh[i] = (_Float16)v;
    if (xf) xf[i] = v;
}

// ---------- mean pool (wave per node) ----------
__global__ void pool_kernel(const float* __restrict__ h, const int* __restrict__ batch,
                            float* __restrict__ pool, float* __restrict__ cnt) {
    int node = (blockIdx.x * blockDim.x + threadIdx.x) >> 5;
    int lane = threadIdx.x & 31;
    if (node >= Nn) return;
    int g = batch[node];
    atomicAdd(pool + (size_t)g * HIDC + lane,      h[(size_t)node * HIDC + lane]);
    atomicAdd(pool + (size_t)g * HIDC + 32 + lane, h[(size_t)node * HIDC + 32 + lane]);
    if (lane == 0) atomicAdd(cnt + g, 1.0f);
}

// ---------- readout: sigmoid(mean @ lin_w + lin_b) ----------
__global__ void readout(const float* __restrict__ pool, const float* __restrict__ cnt,
                        const float* __restrict__ lw, const float* __restrict__ lb,
                        float* __restrict__ out) {
    int g = blockIdx.x * blockDim.x + threadIdx.x;
    if (g >= Gg) return;
    float c = fmaxf(cnt[g], 1.0f);
    float z = lb[0];
    for (int f = 0; f < HIDC; ++f) z += (pool[(size_t)g * HIDC + f] / c) * lw[f];
    out[g] = 1.0f / (1.0f + expf(-z));
}

extern "C" void kernel_launch(void* const* d_in, const int* in_sizes, int n_in,
                              void* d_out, int out_size, void* d_ws, size_t ws_size,
                              hipStream_t stream) {
    const float* x     = (const float*)d_in[0];
    const int*   ei    = (const int*)  d_in[1];
    const float* ew    = (const float*)d_in[2];
    const int*   batch = (const int*)  d_in[3];
    const float* W_[3]  = {(const float*)d_in[4],  (const float*)d_in[10], (const float*)d_in[16]};
    const float* as_[3] = {(const float*)d_in[5],  (const float*)d_in[11], (const float*)d_in[17]};
    const float* ad_[3] = {(const float*)d_in[6],  (const float*)d_in[12], (const float*)d_in[18]};
    const float* We_[3] = {(const float*)d_in[7],  (const float*)d_in[13], (const float*)d_in[19]};
    const float* ae_[3] = {(const float*)d_in[8],  (const float*)d_in[14], (const float*)d_in[20]};
    const float* b_[3]  = {(const float*)d_in[9],  (const float*)d_in[15], (const float*)d_in[21]};
    const float* lw = (const float*)d_in[22];
    const float* lb = (const float*)d_in[23];
    float* out = (float*)d_out;

    // workspace layout
    float* f = (float*)d_ws;
    float* h     = f; f += (size_t)Nn * HIDC;          // GEMM out (f32)
    float* acc   = f; f += (size_t)Nn * HIDC;          // scatter accumulator / layer-3 out
    float* aSrc  = f; f += Nn;
    float* aDst  = f; f += Nn;
    unsigned* mEnc = (unsigned*)f; f += Nn;
    float* sSum  = f; f += Nn;
    float* alpha = f; f += ET;
    float* pool  = f; f += (size_t)Gg * HIDC;
    float* cnt   = f; f += Gg;
    float* scal  = f; f += 4;      // [0..2]=c_e per layer, [3]=sum(edge_weight)
    _Float16* xh128 = (_Float16*)f;                       // layer-1 input, f16 [N x 128]
    _Float16* xh64  = xh128 + (size_t)Nn * FIN;           // layer-2/3 input, f16 [N x 64]
    _Float16* wt1   = xh64  + (size_t)Nn * HIDC;          // Wt f16 [64 x 128]
    _Float16* wt2   = wt1 + (size_t)HIDC * FIN;           // [64 x 64]
    _Float16* wt3   = wt2 + (size_t)HIDC * HIDC;
    _Float16* wt_[3] = {wt1, wt2, wt3};

    const int NB = 256;
    // precompute: scalars, f16 conversions, transposed f16 weights
    fill_f<<<1, 8, 0, stream>>>(scal, 0.f, 4);
    ce_kernel<<<1, HIDC, 0, stream>>>(We_[0], ae_[0], scal + 0);
    ce_kernel<<<1, HIDC, 0, stream>>>(We_[1], ae_[1], scal + 1);
    ce_kernel<<<1, HIDC, 0, stream>>>(We_[2], ae_[2], scal + 2);
    ewsum_kernel<<<(Ee + NB - 1) / NB, NB, 0, stream>>>(ew, scal + 3);
    cvt_f16<<<((Nn * FIN) + NB - 1) / NB, NB, 0, stream>>>(x, xh128, Nn * FIN);
    transpose_w<<<(FIN  * HIDC + NB - 1) / NB, NB, 0, stream>>>(W_[0], wt1, FIN);
    transpose_w<<<(HIDC * HIDC + NB - 1) / NB, NB, 0, stream>>>(W_[1], wt2, HIDC);
    transpose_w<<<(HIDC * HIDC + NB - 1) / NB, NB, 0, stream>>>(W_[2], wt3, HIDC);

    const int relu[3] = {1, 1, 0};
    for (int L = 0; L < 3; ++L) {
        if (L == 0) gemm_wmma<FIN ><<<Nn / 16, 128, 0, stream>>>(xh128, wt_[L], h);
        else        gemm_wmma<HIDC><<<Nn / 16, 128, 0, stream>>>(xh64,  wt_[L], h);
        node_scores<<<(Nn * 32 + NB - 1) / NB, NB, 0, stream>>>(h, as_[L], ad_[L], aSrc, aDst);
        fill_u<<<(Nn + NB - 1) / NB, NB, 0, stream>>>(mEnc, 0x007FFFFFu, Nn); // enc(-inf)
        fill_f<<<(Nn + NB - 1) / NB, NB, 0, stream>>>(sSum, 0.f, Nn);
        fill_f<<<(Nn * HIDC + NB - 1) / NB, NB, 0, stream>>>(acc, 0.f, Nn * HIDC);
        edge_logit_max<<<(ET + NB - 1) / NB, NB, 0, stream>>>(ei, ew, aSrc, aDst,
                                                              scal + L, scal + 3, alpha, mEnc);
        edge_exp_sum<<<(ET + NB - 1) / NB, NB, 0, stream>>>(ei, mEnc, alpha, sSum);
        scatter_edges<<<((size_t)ET * 32 + NB - 1) / NB, NB, 0, stream>>>(ei, h, alpha, sSum, acc);
        if (L < 2)
            finalize<<<(Nn * HIDC + NB - 1) / NB, NB, 0, stream>>>(acc, b_[L], xh64, (float*)nullptr, relu[L]);
        else
            finalize<<<(Nn * HIDC + NB - 1) / NB, NB, 0, stream>>>(acc, b_[L], (_Float16*)nullptr, acc, relu[L]);
    }

    fill_f<<<(Gg * HIDC + NB - 1) / NB, NB, 0, stream>>>(pool, 0.f, Gg * HIDC);
    fill_f<<<(Gg + NB - 1) / NB, NB, 0, stream>>>(cnt, 0.f, Gg);
    pool_kernel<<<(Nn * 32 + NB - 1) / NB, NB, 0, stream>>>(acc, batch, pool, cnt);
    readout<<<(Gg + NB - 1) / NB, NB, 0, stream>>>(pool, cnt, lw, lb, out);
}
